// BasicBlockSystolic_50414326121234
// MI455X (gfx1250) — matrix operand order repaired and checked by reading, not compiled, not run on hardware
//
#include <hip/hip_runtime.h>
#include <hip/hip_bf16.h>

// ---------------------------------------------------------------------------
// ResNet BasicBlock (conv3x3 -> BN -> ReLU -> conv3x3 -> BN -> +id -> ReLU)
// Implicit-GEMM on CDNA5 WMMA (v_wmma_f32_16x16x32_bf16, f32 accumulate),
// double-buffered LDS, A-tiles staged via global_load_async_to_lds_b128,
// B-halo handled via zero-guard page (branch- and select-free staging).
// N=32, C=256, H=W=56.  M = out-channels (256), Npix = 100352,
// K = 256*9 as 9 shifted GEMMs over ci-chunks of 32 (72 K-steps).
// ---------------------------------------------------------------------------

typedef __attribute__((ext_vector_type(16))) __bf16 v16bf;
typedef __attribute__((ext_vector_type(8)))  float  v8f;

#define CCH   256
#define HH    56
#define WWID  56
#define HWSZ  (HH * WWID)          // 3136
#define NIMG  32
#define NPIX  (NIMG * HWSZ)        // 100352

#define BM   128                   // co tile per workgroup
#define BN   128                   // pixel tile per workgroup
#define BK   32                    // ci chunk (WMMA K)
#define LDA  40                    // LDS row stride in halves (32 + pad)
#define LDB  40
#define KSTEPS 72                  // 9 kernel positions * 8 ci chunks
#define ZGUARD_ELEMS (16 * HWSZ)   // covers j*HWSZ, j=0..15

struct __align__(16) U4 { unsigned int x, y, z, w; };

union FragBF {
  U4    q[2];
  v16bf v;
};

// --- CDNA5 async-to-LDS path: builtin takes v4i pointers in AS1/AS3 ---
#if defined(__AMDGCN__)
typedef __attribute__((ext_vector_type(4))) int v4i_t;
typedef __attribute__((address_space(1))) v4i_t as1_v4i;
typedef __attribute__((address_space(3))) v4i_t as3_v4i;
#  if defined(__has_builtin)
#    if __has_builtin(__builtin_amdgcn_global_load_async_to_lds_b128)
#      define ASYNC_LDS 1
#    endif
#    if __has_builtin(__builtin_amdgcn_s_wait_asynccnt)
#      define WAIT_ASYNC() __builtin_amdgcn_s_wait_asynccnt(0)
#    endif
#  endif
#  ifndef ASYNC_LDS
#    define ASYNC_LDS 0
#  endif
#  ifndef WAIT_ASYNC
#    if ASYNC_LDS
#      define WAIT_ASYNC() asm volatile("s_wait_asynccnt 0x0" ::: "memory")
#    else
#      define WAIT_ASYNC()
#    endif
#  endif
#else
#  define ASYNC_LDS 0
#  define WAIT_ASYNC()
#endif

__device__ __forceinline__ unsigned short f2bf(float f) {
  unsigned int u = __float_as_uint(f);
  u += 0x7FFFu + ((u >> 16) & 1u);      // round-to-nearest-even
  return (unsigned short)(u >> 16);
}

// ---------------------------------------------------------------------------
// Prep kernels
// ---------------------------------------------------------------------------

__global__ void prep_scale_bias(const float* __restrict__ g1, const float* __restrict__ b1,
                                const float* __restrict__ m1, const float* __restrict__ v1,
                                const float* __restrict__ g2, const float* __restrict__ b2,
                                const float* __restrict__ m2, const float* __restrict__ v2,
                                float* __restrict__ bias1, float* __restrict__ bias2,
                                float* __restrict__ scale1, float* __restrict__ scale2) {
  int c = threadIdx.x;
  if (c < CCH) {
    float inv1 = g1[c] * rsqrtf(v1[c] + 1e-5f);
    float inv2 = g2[c] * rsqrtf(v2[c] + 1e-5f);
    scale1[c] = inv1;  bias1[c] = b1[c] - m1[c] * inv1;
    scale2[c] = inv2;  bias2[c] = b2[c] - m2[c] * inv2;
  }
}

// w: [co][ci][3][3] fp32  ->  out: [kpos][co][ci] bf16, scaled by BN gamma*rsqrt(var)
__global__ void prep_weights(const float* __restrict__ w,
                             const float* __restrict__ scale,
                             unsigned short* __restrict__ out) {
  int o = blockIdx.x * blockDim.x + threadIdx.x;
  if (o >= 9 * CCH * CCH) return;
  int kpos = o >> 16;
  int rem  = o & 0xFFFF;
  int co   = rem >> 8;
  int ci   = rem & 0xFF;
  float val = w[((long)co * CCH + ci) * 9 + kpos] * scale[co];
  out[o] = f2bf(val);
}

__global__ void prep_x(const float* __restrict__ x, unsigned short* __restrict__ xb, int n) {
  int i = blockIdx.x * blockDim.x + threadIdx.x;
  if (i < n) xb[i] = f2bf(x[i]);
}

__global__ void zero_guard(unsigned short* __restrict__ z, int n) {
  int i = blockIdx.x * blockDim.x + threadIdx.x;
  if (i < n) z[i] = 0;
}

// ---------------------------------------------------------------------------
// Implicit-GEMM conv3x3 (+BN-bias) with WMMA.
//   FUSE_RES = false : out = bf16( relu(conv + bias) )      -> Ybf
//   FUSE_RES = true  : out = f32 ( relu(conv + bias + id) ) -> Yf (nontemporal)
// Block: 128 co x 128 pix, 8 waves; wave tile: 32 co x 64 pix = 2x4 WMMA tiles.
// ---------------------------------------------------------------------------

template <bool FUSE_RES>
__global__ __launch_bounds__(256)
void conv3x3_bnrelu_wmma(const unsigned short* __restrict__ Xb,   // bf16 [N][C][H][W]
                         const unsigned short* __restrict__ Wb,   // bf16 [9][co][ci]
                         const float* __restrict__ bias,          // [C]
                         const float* __restrict__ resid,         // fp32 or null
                         const unsigned short* __restrict__ Zg,   // zeroed guard strip
                         unsigned short* __restrict__ Ybf,
                         float* __restrict__ Yf) {
  __shared__ __align__(16) unsigned short As[2][BM * LDA];
  __shared__ __align__(16) unsigned short Bs[2][BN * LDB];

  const int t    = threadIdx.x;
  const int lane = t & 31;
  const int wv   = t >> 5;
  const int waveM = (wv & 3) * 32;          // 2 m-tiles of 16
  const int waveN = (wv >> 2) * 64;         // 4 n-tiles of 16
  const int blockM   = blockIdx.x * BM;
  const int blockPix = blockIdx.y * BN;

  // B staging role: one pixel, 16 contiguous channels
  const int pB  = t & 127;
  const int cB0 = (t >> 7) * 16;
  const int gpB = blockPix + pB;
  const int nimgB = gpB / HWSZ;
  const int hwB   = gpB - nimgB * HWSZ;
  const int hB    = hwB / WWID;
  const int wB    = hwB - hB * WWID;
  const long baseB = (long)nimgB * CCH * HWSZ;

  // A staging role: half a weight row (16 halves)
  const int rowA = t >> 1;
  const int hsA  = (t & 1) * 16;

  v8f acc[2][4] = {};

  const int lm = lane & 15;
  const int hi = lane >> 4;

  // -------- tile stage (async A copy + guard-paged B gather) --------
  auto stage = [&](int s, int bf) {
    const int kpos = s >> 3;
    const int ck   = s & 7;
    const int ci0  = ck * BK;
    const int kh   = kpos / 3 - 1;
    const int kw   = kpos - (kpos / 3) * 3 - 1;

    // A tile: 128 co x 32 ci, contiguous 32B per thread
    const unsigned short* gA =
        Wb + ((long)kpos * CCH + blockM + rowA) * CCH + ci0 + hsA;
    unsigned short* lA = &As[bf][rowA * LDA + hsA];
#if ASYNC_LDS
    __builtin_amdgcn_global_load_async_to_lds_b128((as1_v4i*)gA, (as3_v4i*)lA, 0, 0);
    __builtin_amdgcn_global_load_async_to_lds_b128((as1_v4i*)gA, (as3_v4i*)lA, 16, 0);
#else
    *(U4*)lA       = *(const U4*)gA;
    *(U4*)(lA + 8) = *(const U4*)(gA + 8);
#endif

    // B tile: 128 pix x 32 ci; OOB pixels read a zeroed guard strip instead
    // of per-element selects (single 64-bit base select).
    const int hs  = hB + kh;
    const int ws2 = wB + kw;
    const bool ok = ((unsigned)hs < (unsigned)HH) & ((unsigned)ws2 < (unsigned)WWID);
    const long off0 = baseB + (long)(ci0 + cB0) * HWSZ + (long)hs * WWID + ws2;
    const unsigned short* src = ok ? (Xb + off0) : Zg;
    union { unsigned short u[16]; U4 q[2]; } pk;
#pragma unroll
    for (int j = 0; j < 16; ++j) pk.u[j] = src[(long)j * HWSZ];
    U4* lB = (U4*)&Bs[bf][pB * LDB + cB0];
    lB[0] = pk.q[0];
    lB[1] = pk.q[1];
  };

  // -------- compute one K-step from buffer `cur` --------
  auto compute = [&](int cur) {
    FragBF a[2], b[4];
#pragma unroll
    for (int mt = 0; mt < 2; ++mt) {
      // A 16x32: lane<16 -> K 0..15 ; lane>=16 -> K 16..31
      const unsigned short* rp = &As[cur][(waveM + mt * 16 + lm) * LDA + hi * 16];
      a[mt].q[0] = *(const U4*)(rp);
      a[mt].q[1] = *(const U4*)(rp + 8);
    }
#pragma unroll
    for (int nt = 0; nt < 4; ++nt) {
      // B 32x16: lane<16 -> K 0..15 ; lane>=16 -> K 16..31
      const unsigned short* cp = &Bs[cur][(waveN + nt * 16 + lm) * LDB + hi * 16];
      b[nt].q[0] = *(const U4*)(cp);
      b[nt].q[1] = *(const U4*)(cp + 8);
    }
#pragma unroll
    for (int mt = 0; mt < 2; ++mt)
#pragma unroll
      for (int nt = 0; nt < 4; ++nt)
        acc[mt][nt] = __builtin_amdgcn_wmma_f32_16x16x32_bf16(
            false, a[mt].v, false, b[nt].v, (short)0, acc[mt][nt], false, false);
  };

  // -------- pipelined main loop (single barrier per K-step) --------
  stage(0, 0);
  WAIT_ASYNC();
  __syncthreads();
  for (int s = 0; s < KSTEPS; ++s) {
    const int cur = s & 1;
    if (s + 1 < KSTEPS) stage(s + 1, cur ^ 1);
    compute(cur);
    WAIT_ASYNC();
    __syncthreads();
  }

  // -------- epilogue: C/D layout lane<16 -> M=r, lane>=16 -> M=8+r --------
#pragma unroll
  for (int mt = 0; mt < 2; ++mt) {
#pragma unroll
    for (int nt = 0; nt < 4; ++nt) {
      const int n  = waveN + nt * 16 + lm;
      const int gp = blockPix + n;
      const int nimg = gp / HWSZ;
      const int hw   = gp - nimg * HWSZ;
#pragma unroll
      for (int r = 0; r < 8; ++r) {
        const int m  = waveM + mt * 16 + hi * 8 + r;
        const int co = blockM + m;
        const long idx = ((long)nimg * CCH + co) * HWSZ + hw;
        float v = acc[mt][nt][r] + bias[co];
        if (FUSE_RES) {
          v += resid[idx];
          v = v > 0.0f ? v : 0.0f;
          __builtin_nontemporal_store(v, &Yf[idx]);   // final output: bypass-friendly
        } else {
          v = v > 0.0f ? v : 0.0f;
          Ybf[idx] = f2bf(v);                          // re-read by conv2: keep RT
        }
      }
    }
  }
}

// ---------------------------------------------------------------------------
// Host launcher
// ---------------------------------------------------------------------------

extern "C" void kernel_launch(void* const* d_in, const int* in_sizes, int n_in,
                              void* d_out, int out_size, void* d_ws, size_t ws_size,
                              hipStream_t stream) {
  const float* x  = (const float*)d_in[0];
  const float* w1 = (const float*)d_in[1];
  const float* g1 = (const float*)d_in[2];
  const float* b1 = (const float*)d_in[3];
  const float* m1 = (const float*)d_in[4];
  const float* v1 = (const float*)d_in[5];
  const float* w2 = (const float*)d_in[6];
  const float* g2 = (const float*)d_in[7];
  const float* b2 = (const float*)d_in[8];
  const float* m2 = (const float*)d_in[9];
  const float* v2 = (const float*)d_in[10];

  char* ws = (char*)d_ws;
  size_t off = 0;
  auto take = [&](size_t bytes) {
    void* p = ws + off;
    off += (bytes + 255) & ~size_t(255);
    return p;
  };
  unsigned short* W1b   = (unsigned short*)take(9 * CCH * CCH * 2);
  unsigned short* W2b   = (unsigned short*)take(9 * CCH * CCH * 2);
  float*          bias1 = (float*)take(CCH * 4);
  float*          bias2 = (float*)take(CCH * 4);
  float*          sc1   = (float*)take(CCH * 4);
  float*          sc2   = (float*)take(CCH * 4);
  unsigned short* Zg    = (unsigned short*)take((size_t)ZGUARD_ELEMS * 2);
  unsigned short* Xb    = (unsigned short*)take((size_t)NPIX * CCH * 2);
  unsigned short* Y1b   = (unsigned short*)take((size_t)NPIX * CCH * 2);
  (void)ws_size; (void)in_sizes; (void)n_in; (void)out_size;

  prep_scale_bias<<<1, 256, 0, stream>>>(g1, b1, m1, v1, g2, b2, m2, v2,
                                         bias1, bias2, sc1, sc2);

  const int wElems = 9 * CCH * CCH;
  prep_weights<<<(wElems + 255) / 256, 256, 0, stream>>>(w1, sc1, W1b);
  prep_weights<<<(wElems + 255) / 256, 256, 0, stream>>>(w2, sc2, W2b);

  zero_guard<<<(ZGUARD_ELEMS + 255) / 256, 256, 0, stream>>>(Zg, ZGUARD_ELEMS);

  const int xElems = NPIX * CCH;
  prep_x<<<(xElems + 255) / 256, 256, 0, stream>>>(x, Xb, xElems);

  dim3 grid(CCH / BM, NPIX / BN);  // (2, 784)
  conv3x3_bnrelu_wmma<false><<<grid, 256, 0, stream>>>(Xb, W1b, bias1, nullptr, Zg, Y1b, nullptr);
  conv3x3_bnrelu_wmma<true><<<grid, 256, 0, stream>>>(Y1b, W2b, bias2, x, Zg, nullptr, (float*)d_out);
}
